// MultiGalerkinLayer_46505905881706
// MI455X (gfx1250) — compile-verified, matrix operand order checked by reading, not run on hardware
//
#include <hip/hip_runtime.h>
#include <hip/hip_bf16.h>
#include <math.h>

// ---------------------------------------------------------------------------
// MultiGalerkinLayer for MI455X (gfx1250, wave32).
// Heavy einsums / 1x1 convs -> bf16 WMMA (v_wmma_f32_16x16x32_bf16) GEMM with
// 64x256 block tiles, 8 WMMA per wave per K-step, all-vector LDS staging
// (B operands stored pre-transposed [N][K]), split-K for the projection GEMM.
// K=3 convs / restrictions / prolongation / gelu -> VALU+LDS kernels.
// ---------------------------------------------------------------------------

typedef __bf16 bf16;
typedef __attribute__((ext_vector_type(16))) __bf16 v16bf;
typedef __attribute__((ext_vector_type(8)))  __bf16 bf16x8;
typedef __attribute__((ext_vector_type(8)))  float  v8f;

__device__ __forceinline__ bf16 f2bf(float f) {
  union { float f; unsigned u; } v; v.f = f;
  unsigned r = v.u + 0x7FFFu + ((v.u >> 16) & 1u);   // round-to-nearest-even
  union { unsigned short s; bf16 b; } o; o.s = (unsigned short)(r >> 16);
  return o.b;
}

#define BM 64
#define BN 256
#define BK 32

// C[M,N] = A[M,K] * BT[N,K]^T (+C if acc).  bf16 in, f32 out.
// M multiple of 64, N multiple of 256 (grid-implied), K multiple of 32.
// All row strides multiples of 8 elements -> 16B-aligned vector staging.
// gridDim = (N/256, M/64, batch*SK). Split-K: chunk writes C + chunk*sCk.
__global__ __launch_bounds__(256)
void wmma_gemm_bt(const bf16* __restrict__ Ag, const bf16* __restrict__ BTg,
                  float* __restrict__ Cg,
                  int K, int lda, int ldbt, int ldc,
                  long sA, long sB, long sC, long sCk, int SK, int acc)
{
  __shared__ __align__(16) bf16 As[BM][BK + 8];
  __shared__ __align__(16) bf16 Bs[BN][BK + 8];   // [n][k]

  const int chunk = blockIdx.z % SK;
  const int batch = blockIdx.z / SK;

  const bf16* A  = Ag  + (long)batch * sA;
  const bf16* BT = BTg + (long)batch * sB;
  float*      C  = Cg  + (long)batch * sC + (long)chunk * sCk;

  const int m0 = blockIdx.y * BM;
  const int n0 = blockIdx.x * BN;

  const int tid  = threadIdx.x;
  const int lane = tid & 31;
  const int wid  = tid >> 5;       // 0..7 (8 waves)
  const int wrow = wid & 1;        // 32-row half of BM
  const int wcol = wid >> 1;       // 64-col quarter of BN

  const int frow = lane & 15;
  const int fseg = (lane >> 4) * 8;   // fragment K-segment: 0 or 8

  // staging coordinates
  const int ar = tid >> 2;            // A row 0..63
  const int as = (tid & 3) * 8;       // k-seg {0,8,16,24}
  const bf16* Arow  = A  + (long)(m0 + ar) * lda + as;
  const bf16* BTrow = BT + (long)(n0 + ar) * ldbt + as;   // rows ar, ar+64, ...

  // split-K bounds (multiples of BK)
  const int steps = K / BK;
  const int cs    = (steps + SK - 1) / SK;
  int k0   = chunk * cs * BK;
  int kEnd = (chunk + 1) * cs * BK;
  if (kEnd > K) kEnd = K;

  v8f acc_[2][4];
#pragma unroll
  for (int i = 0; i < 2; ++i)
#pragma unroll
    for (int j = 0; j < 4; ++j) acc_[i][j] = (v8f){};

  for (; k0 < kEnd; k0 += BK) {
    bf16x8 av = *(const bf16x8*)(Arow + k0);
    bf16x8 bv0 = *(const bf16x8*)(BTrow + k0);
    bf16x8 bv1 = *(const bf16x8*)(BTrow + (long)64 * ldbt + k0);
    bf16x8 bv2 = *(const bf16x8*)(BTrow + (long)128 * ldbt + k0);
    bf16x8 bv3 = *(const bf16x8*)(BTrow + (long)192 * ldbt + k0);

    if (k0 + BK < kEnd) {  // near-cache prefetch of next tiles
      __builtin_prefetch(Arow + k0 + BK, 0, 3);
      __builtin_prefetch(BTrow + k0 + BK, 0, 3);
      __builtin_prefetch(BTrow + (long)128 * ldbt + k0 + BK, 0, 3);
    }

    *(bf16x8*)&As[ar][as] = av;
    *(bf16x8*)&Bs[ar][as]       = bv0;
    *(bf16x8*)&Bs[ar + 64][as]  = bv1;
    *(bf16x8*)&Bs[ar + 128][as] = bv2;
    *(bf16x8*)&Bs[ar + 192][as] = bv3;
    __syncthreads();

    v16bf afr[2], bfr[4];
#pragma unroll
    for (int i = 0; i < 2; ++i) {
      bf16x8 lo = *(const bf16x8*)&As[wrow * 32 + i * 16 + frow][fseg];
      bf16x8 hi = *(const bf16x8*)&As[wrow * 32 + i * 16 + frow][fseg + 16];
#pragma unroll
      for (int e = 0; e < 8; ++e) { afr[i][e] = lo[e]; afr[i][e + 8] = hi[e]; }
    }
#pragma unroll
    for (int j = 0; j < 4; ++j) {
      bf16x8 lo = *(const bf16x8*)&Bs[wcol * 64 + j * 16 + frow][fseg];
      bf16x8 hi = *(const bf16x8*)&Bs[wcol * 64 + j * 16 + frow][fseg + 16];
#pragma unroll
      for (int e = 0; e < 8; ++e) { bfr[j][e] = lo[e]; bfr[j][e + 8] = hi[e]; }
    }

#pragma unroll
    for (int i = 0; i < 2; ++i)
#pragma unroll
      for (int j = 0; j < 4; ++j)
        acc_[i][j] = __builtin_amdgcn_wmma_f32_16x16x32_bf16(
            false, afr[i], false, bfr[j], (short)0, acc_[i][j], false, false);

    __syncthreads();
  }

  // C/D layout: VGPR r -> M=r (lanes 0-15), M=8+r (lanes 16-31); N = lane%16
  const int rowB = m0 + wrow * 32 + ((lane >> 4) ? 8 : 0);
  const int colB = n0 + wcol * 64 + (lane & 15);
#pragma unroll
  for (int i = 0; i < 2; ++i)
#pragma unroll
    for (int j = 0; j < 4; ++j) {
#pragma unroll
      for (int r = 0; r < 8; ++r) {
        long idx = (long)(rowB + i * 16 + r) * ldc + (colB + j * 16);
        C[idx] = acc ? (C[idx] + acc_[i][j][r]) : acc_[i][j][r];
      }
    }
}

// ------------------------- elementwise / small kernels ----------------------

__global__ void k_convert_bf16(const float* __restrict__ src, bf16* __restrict__ dst, long n) {
  long i = (long)blockIdx.x * blockDim.x + threadIdx.x;
  if (i < n) dst[i] = f2bf(src[i]);
}

// split-K reduction: out[i] = sum_s part[s*stride + i]
__global__ void k_reduce_sk(const float* __restrict__ part, float* __restrict__ out,
                            long n, int sk) {
  long i = (long)blockIdx.x * blockDim.x + threadIdx.x;
  if (i >= n) return;
  float acc = 0.f;
  for (int s = 0; s < sk; ++s) acc += part[(long)s * n + i];
  out[i] = acc;
}

// x = concat(a,u): channel-major (GEMM A) and position-major (GEMM B^T) packs
__global__ void k_build_x(const float* __restrict__ a, const float* __restrict__ u,
                          bf16* __restrict__ xcm, bf16* __restrict__ xpm,
                          int n, int np) {
  int p  = blockIdx.x * blockDim.x + threadIdx.x;
  int bc = blockIdx.y;
  if (p >= np) return;
  int c = bc & 127, b = bc >> 7;
  float v = 0.f;
  if (p < n) v = (c < 64) ? a[((long)b * 64 + c) * n + p]
                          : u[((long)b * 64 + (c - 64)) * n + p];
  bf16 bv = f2bf(v);
  xcm[(long)bc * np + p] = bv;
  xpm[((long)b * np + p) * 128 + c] = bv;
}

// pack [np,256] bf16 (zero rows p>=n) -- natural layout of bases/wbases
__global__ void k_pack_rows(const float* __restrict__ src, bf16* __restrict__ dst,
                            int n, long total) {
  long i = (long)blockIdx.x * blockDim.x + threadIdx.x;
  if (i >= total) return;
  long p = i >> 8;
  dst[i] = (p < n) ? f2bf(src[i]) : f2bf(0.f);
}

// bT[k][p] = bases[p][k], padded np (zero for p>=n); grid.y = 256 (k)
__global__ void k_transpose_b(const float* __restrict__ bases, bf16* __restrict__ bT,
                              int n, int np) {
  int p = blockIdx.x * blockDim.x + threadIdx.x;
  int k = blockIdx.y;
  if (p >= np) return;
  bT[(long)k * np + p] = (p < n) ? f2bf(bases[(long)p * 256 + k]) : f2bf(0.f);
}

// xh2[b,o,k] = sum_i xh[b,i,k] * spw[i,o,k]   (einsum 'bik,iok->bok')
__global__ void k_mode_mul(const float* __restrict__ xh, const float* __restrict__ spw,
                           float* __restrict__ xh2) {
  long i = (long)blockIdx.x * blockDim.x + threadIdx.x;
  if (i >= 16L * 128 * 256) return;
  int k = (int)(i & 255);
  int o = (int)((i >> 8) & 127);
  long b = i >> 15;
  float acc = 0.f;
#pragma unroll 4
  for (int ci = 0; ci < 128; ++ci)
    acc += xh[(b * 128 + ci) * 256 + k] * spw[((long)ci * 128 + o) * 256 + k];
  xh2[i] = acc;
}

// t(position-major) = bf16( x + gelu(x1 + w_b[c]) )
__global__ void k_gelu_combine(const float* __restrict__ x1,
                               const float* __restrict__ a, const float* __restrict__ u,
                               const float* __restrict__ wb,
                               bf16* __restrict__ tpm, int n, int np) {
  int p  = blockIdx.x * blockDim.x + threadIdx.x;
  int bc = blockIdx.y;
  if (p >= np) return;
  int c = bc & 127, b = bc >> 7;
  bf16 outv = f2bf(0.f);
  if (p < n) {
    float s = x1[(long)bc * np + p] + wb[c];
    float g = 0.5f * s * (1.f + erff(s * 0.70710678118654752f));
    float xc = (c < 64) ? a[((long)b * 64 + c) * n + p]
                        : u[((long)b * 64 + (c - 64)) * n + p];
    outv = f2bf(xc + g);
  }
  tpm[((long)b * np + p) * 128 + c] = outv;
}

// u[b,o,p] += bias[o] + sum_c sum_t sw[o,c,t] * res[b,c,p-1+t],
// res = f - pos - fc_b[c]; pos has padded stride np.
__global__ __launch_bounds__(256)
void k_smooth(float* __restrict__ u, const float* __restrict__ f,
              const float* __restrict__ pos, const float* __restrict__ fcb,
              const float* __restrict__ sw, const float* __restrict__ sb,
              int n, int np) {
  __shared__ float res[64][66];
  int b  = blockIdx.y;
  int p0 = blockIdx.x * 64;
  int tid = threadIdx.x;

  for (int idx = tid; idx < 64 * 66; idx += 256) {
    int c = idx / 66, pp = idx % 66;
    int p = p0 + pp - 1;
    float v = 0.f;
    if (p >= 0 && p < n)
      v = f[((long)b * 64 + c) * n + p] - pos[((long)b * 64 + c) * np + p] - fcb[c];
    res[c][pp] = v;
  }
  __syncthreads();

  int pl = tid & 63;
  int og = tid >> 6;          // 0..3
  if (p0 + pl >= n) return;
#pragma unroll
  for (int i = 0; i < 16; ++i) {
    int o = og * 16 + i;
    float acc = sb[o];
    const float* wrow = &sw[(long)o * 64 * 3];
#pragma unroll 4
    for (int c = 0; c < 64; ++c) {
      const float* w = &wrow[c * 3];
      acc += w[0] * res[c][pl] + w[1] * res[c][pl + 1] + w[2] * res[c][pl + 2];
    }
    u[((long)b * 64 + o) * n + (p0 + pl)] += acc;
  }
}

// y[b,o,j] = sum_c sum_t R[o,c,t] * x[b,c,2j-1+t]; grid.y = B*64
__global__ void k_restrict(const float* __restrict__ x, float* __restrict__ y,
                           const float* __restrict__ R, int n, int m) {
  int j  = blockIdx.x * blockDim.x + threadIdx.x;
  int bo = blockIdx.y;
  if (j >= m) return;
  int o = bo & 63, b = bo >> 6;
  float acc = 0.f;
  for (int c = 0; c < 64; ++c) {
    const float* w  = &R[(o * 64 + c) * 3];
    const float* xr = &x[((long)b * 64 + c) * n];
#pragma unroll
    for (int t = 0; t < 3; ++t) {
      int p = 2 * j - 1 + t;
      if (p >= 0 && p < n) acc += w[t] * xr[p];
    }
  }
  y[(long)bo * m + j] = acc;
}

// bases restriction: out[j,k] = sum_t Rb[t] * in[2j-1+t, k]
__global__ void k_restrict_bases(const float* __restrict__ bin, float* __restrict__ bout,
                                 const float* __restrict__ Rb, int n, long total) {
  long i = (long)blockIdx.x * blockDim.x + threadIdx.x;
  if (i >= total) return;
  long k = i & 255;
  long j = i >> 8;
  float acc = 0.f;
#pragma unroll
  for (int t = 0; t < 3; ++t) {
    long p = 2 * j - 1 + t;
    if (p >= 0 && p < n) acc += Rb[t] * bin[p * 256 + k];
  }
  bout[i] = acc;
}

// out[b,o,p] = base[b,o,p] + convT(x, w)[b,o,p]; grid.y = B*64
__global__ void k_prolong_add(const float* __restrict__ base, const float* __restrict__ x,
                              const float* __restrict__ w, float* __restrict__ out,
                              int L, int nOut) {
  int p  = blockIdx.x * blockDim.x + threadIdx.x;
  int bo = blockIdx.y;
  if (p >= nOut) return;
  int o = bo & 63, b = bo >> 6;
  float acc = base[(long)bo * nOut + p];
  if ((p & 1) == 0) {
    int j = p >> 1;
    for (int c = 0; c < 64; ++c)
      acc += w[((long)c * 64 + o) * 3 + 1] * x[((long)b * 64 + c) * L + j];
  } else {
    int j0 = (p - 1) >> 1, j1 = j0 + 1;
    for (int c = 0; c < 64; ++c) {
      const float* xr = &x[((long)b * 64 + c) * L];
      acc += w[((long)c * 64 + o) * 3 + 2] * xr[j0];
      if (j1 < L) acc += w[((long)c * 64 + o) * 3 + 0] * xr[j1];
    }
  }
  out[(long)bo * nOut + p] = acc;
}

// ------------------------------- host side ---------------------------------

static inline long cdivl(long a, long b) { return (a + b - 1) / b; }

extern "C" void kernel_launch(void* const* d_in, const int* in_sizes, int n_in,
                              void* d_out, int out_size, void* d_ws, size_t ws_size,
                              hipStream_t stream) {
  (void)in_sizes; (void)n_in; (void)out_size; (void)ws_size;
  const int Bn = 16;
  const int N0 = 8193, N1 = 4097, N2 = 2049;
  const int P0 = 8448, P1 = 4352, P2 = 2304;    // ceil(n/256)*256
  const int SPLITK = 8;

  const float* a_in   = (const float*)d_in[0];
  const float* u_in   = (const float*)d_in[1];
  const float* f_in   = (const float*)d_in[2];
  const float* bases0 = (const float*)d_in[3];
  const float* wbas0  = (const float*)d_in[4];
  const float* sp_w   = (const float*)d_in[5];
  const float* w_w    = (const float*)d_in[6];
  const float* w_b    = (const float*)d_in[7];
  const float* fc_w   = (const float*)d_in[8];
  const float* fc_b   = (const float*)d_in[9];
  const float* Ra     = (const float*)d_in[10];
  const float* Ru     = (const float*)d_in[11];
  const float* Rf     = (const float*)d_in[12];
  const float* Rb     = (const float*)d_in[13];
  const float* Rwb    = (const float*)d_in[14];
  const float* sm_w   = (const float*)d_in[15];
  const float* sm_b   = (const float*)d_in[16];
  const float* prol_w = (const float*)d_in[17];
  float* outp = (float*)d_out;

  char* wsp = (char*)d_ws;
  auto alloc = [&](size_t bytes) -> void* {
    void* p = (void*)wsp;
    wsp += (bytes + 255) & ~(size_t)255;
    return p;
  };

  float* u0  = (float*)alloc(sizeof(float) * (size_t)Bn * 64 * N0);
  float* u1  = (float*)alloc(sizeof(float) * (size_t)Bn * 64 * N1);
  float* u2  = (float*)alloc(sizeof(float) * (size_t)Bn * 64 * N2);
  float* a1  = (float*)alloc(sizeof(float) * (size_t)Bn * 64 * N1);
  float* a2  = (float*)alloc(sizeof(float) * (size_t)Bn * 64 * N2);
  float* f1  = (float*)alloc(sizeof(float) * (size_t)Bn * 64 * N1);
  float* f2  = (float*)alloc(sizeof(float) * (size_t)Bn * 64 * N2);
  float* up1 = (float*)alloc(sizeof(float) * (size_t)Bn * 64 * N1);
  float* b1  = (float*)alloc(sizeof(float) * (size_t)N1 * 256);
  float* wb1 = (float*)alloc(sizeof(float) * (size_t)N1 * 256);
  float* b2  = (float*)alloc(sizeof(float) * (size_t)N2 * 256);
  float* wb2 = (float*)alloc(sizeof(float) * (size_t)N2 * 256);
  bf16* x_cm   = (bf16*)alloc(sizeof(bf16) * (size_t)Bn * 128 * P0);
  bf16* x_pm   = (bf16*)alloc(sizeof(bf16) * (size_t)Bn * 128 * P0);
  bf16* wbT_bf = (bf16*)alloc(sizeof(bf16) * (size_t)256 * P0);   // proj  B^T
  bf16* b_pack = (bf16*)alloc(sizeof(bf16) * (size_t)P0 * 256);   // recon B^T
  float* xh_part = (float*)alloc(sizeof(float) * (size_t)SPLITK * Bn * 128 * 256);
  float* xh    = (float*)alloc(sizeof(float) * (size_t)Bn * 128 * 256);
  float* xh2   = (float*)alloc(sizeof(float) * (size_t)Bn * 128 * 256);
  bf16* xh2_bf = (bf16*)alloc(sizeof(bf16) * (size_t)Bn * 128 * 256);
  float* x1    = (float*)alloc(sizeof(float) * (size_t)Bn * 128 * P0);
  bf16* t_pm   = (bf16*)alloc(sizeof(bf16) * (size_t)Bn * 128 * P0);
  float* pos   = (float*)alloc(sizeof(float) * (size_t)Bn * 64 * P0);
  bf16* w_w_bf  = (bf16*)alloc(sizeof(bf16) * 128 * 128);
  bf16* fc_w_bf = (bf16*)alloc(sizeof(bf16) * 64 * 128);

  const int TPB = 256;
  auto cvt = [&](const float* s, bf16* d, long n) {
    k_convert_bf16<<<cdivl(n, TPB), TPB, 0, stream>>>(s, d, n);
  };

  cvt(w_w, w_w_bf, 128 * 128);
  cvt(fc_w, fc_w_bf, 64 * 128);
  hipMemcpyAsync(u0, u_in, sizeof(float) * (size_t)Bn * 64 * N0,
                 hipMemcpyDeviceToDevice, stream);

  // one "positive()" pass -> pos[B,64,np]
  auto positive = [&](const float* aL, const float* uL, int n, int np) {
    dim3 gx(cdivl(np, TPB), Bn * 128);
    k_build_x<<<gx, TPB, 0, stream>>>(aL, uL, x_cm, x_pm, n, np);
    // proj (split-K): xh = x * wb   [128 x np]*[np x 256]
    wmma_gemm_bt<<<dim3(1, 2, Bn * SPLITK), TPB, 0, stream>>>(
        x_cm, wbT_bf, xh_part, np, np, np, 256,
        (long)128 * np, 0L, 32768L, (long)Bn * 32768L, SPLITK, 0);
    k_reduce_sk<<<cdivl(16L * 128 * 256, TPB), TPB, 0, stream>>>(
        xh_part, xh, 16L * 128 * 256, SPLITK);
    k_mode_mul<<<cdivl(16L * 128 * 256, TPB), TPB, 0, stream>>>(xh, sp_w, xh2);
    cvt(xh2, xh2_bf, 16L * 128 * 256);
    // recon: x1 = xh2 * b^T  [128 x 256]*[256 x np]
    wmma_gemm_bt<<<dim3(np / BN, 2, Bn), TPB, 0, stream>>>(
        xh2_bf, b_pack, x1, 256, 256, 256, np,
        32768L, 0L, (long)128 * np, 0L, 1, 0);
    // pointwise: x1 += w_w * x  [128 x 128]*[128 x np]
    wmma_gemm_bt<<<dim3(np / BN, 2, Bn), TPB, 0, stream>>>(
        w_w_bf, x_pm, x1, 128, 128, 128, np,
        0L, (long)np * 128, (long)128 * np, 0L, 1, 1);
    k_gelu_combine<<<gx, TPB, 0, stream>>>(x1, aL, uL, w_b, t_pm, n, np);
    // fc: pos = fc_w * t  [64 x 128]*[128 x np]
    wmma_gemm_bt<<<dim3(np / BN, 1, Bn), TPB, 0, stream>>>(
        fc_w_bf, t_pm, pos, 128, 128, 128, np,
        0L, (long)np * 128, (long)64 * np, 0L, 1, 0);
  };

  auto do_level = [&](const float* aL, float* uL, const float* fL,
                      const float* bL, const float* wbL, int n, int np, int level) {
    k_pack_rows<<<cdivl((long)np * 256, TPB), TPB, 0, stream>>>(bL, b_pack, n, (long)np * 256);
    k_transpose_b<<<dim3(cdivl(np, TPB), 256), TPB, 0, stream>>>(wbL, wbT_bf, n, np);
    for (int s = 0; s < 2; ++s) {
      positive(aL, uL, n, np);
      k_smooth<<<dim3(cdivl(n, 64), Bn), TPB, 0, stream>>>(
          uL, fL, pos, fc_b,
          sm_w + (long)(level * 2 + s) * 64 * 64 * 3,
          sm_b + (long)(level * 2 + s) * 64, n, np);
    }
  };

  // ---- level 0 ----
  do_level(a_in, u0, f_in, bases0, wbas0, N0, P0, 0);
  {
    dim3 g(cdivl(N1, TPB), Bn * 64);
    k_restrict<<<g, TPB, 0, stream>>>(a_in, a1, Ra, N0, N1);
    k_restrict<<<g, TPB, 0, stream>>>(u0,  u1, Ru, N0, N1);
    k_restrict<<<g, TPB, 0, stream>>>(f_in, f1, Rf, N0, N1);
    long tbm = (long)N1 * 256;
    k_restrict_bases<<<cdivl(tbm, TPB), TPB, 0, stream>>>(bases0, b1, Rb,  N0, tbm);
    k_restrict_bases<<<cdivl(tbm, TPB), TPB, 0, stream>>>(wbas0, wb1, Rwb, N0, tbm);
  }
  // ---- level 1 ----
  do_level(a1, u1, f1, b1, wb1, N1, P1, 1);
  {
    dim3 g(cdivl(N2, TPB), Bn * 64);
    k_restrict<<<g, TPB, 0, stream>>>(a1, a2, Ra + 64L * 64 * 3, N1, N2);
    k_restrict<<<g, TPB, 0, stream>>>(u1, u2, Ru + 64L * 64 * 3, N1, N2);
    k_restrict<<<g, TPB, 0, stream>>>(f1, f2, Rf + 64L * 64 * 3, N1, N2);
    long tbm = (long)N2 * 256;
    k_restrict_bases<<<cdivl(tbm, TPB), TPB, 0, stream>>>(b1,  b2,  Rb + 3,  N1, tbm);
    k_restrict_bases<<<cdivl(tbm, TPB), TPB, 0, stream>>>(wb1, wb2, Rwb + 3, N1, tbm);
  }
  // ---- level 2 (coarsest) ----
  do_level(a2, u2, f2, b2, wb2, N2, P2, 2);

  // ---- prolongation ----
  {
    k_prolong_add<<<dim3(cdivl(N1, TPB), Bn * 64), TPB, 0, stream>>>(
        u1, u2, prol_w + 64L * 64 * 3, up1, N2, N1);
    k_prolong_add<<<dim3(cdivl(N0, TPB), Bn * 64), TPB, 0, stream>>>(
        u0, up1, prol_w, outp, N1, N0);
  }
}
